// MultiHeadAttention_25872882992056
// MI455X (gfx1250) — matrix operand order repaired and checked by reading, not compiled, not run on hardware
//
#include <hip/hip_runtime.h>

// MI455X / gfx1250: wave32, WMMA 16x16x32 bf16 with f32 accumulation.
// Compute-bound MHA (~155 GFLOP vs ~200MB HBM traffic -> WMMA path).
// Attention uses 64-key steps so softmax shuffle reductions amortize; GEMMs
// load all fragments per k-step into distinct regs for load/WMMA overlap.

typedef __attribute__((ext_vector_type(16))) __bf16 v16bf;
typedef __attribute__((ext_vector_type(8)))  float  v8f;

#define D_MODEL 1024
#define N_HEADS 16
#define HEAD_DIM 64
#define BATCH 4
#define SEQ 2048
#define M_TOTAL (BATCH * SEQ)   // 8192 tokens

#define WMMA_BF16(a, b, c) \
  __builtin_amdgcn_wmma_f32_16x16x32_bf16(false, (a), false, (b), (short)0, (c), false, false)

__device__ __forceinline__ unsigned short f2bf(float f) {
  unsigned int u = __float_as_uint(f);
  u += 0x7FFFu + ((u >> 16) & 1u);       // round-to-nearest-even
  return (unsigned short)(u >> 16);
}

union FragU { v16bf v; uint4 q[2]; };

// A-fragment 16x32 bf16 from row-major [m][k] tile (lda in elements).
__device__ __forceinline__ v16bf load_a_frag(const unsigned short* base, int lda,
                                             int kb, int lane) {
  const int m = lane & 15, h = lane >> 4;
  FragU u;
  const unsigned short* p = base + (size_t)m * lda + kb + h * 8;
  u.q[0] = *(const uint4*)(p);
  u.q[1] = *(const uint4*)(p + 16);
  return u.v;
}

// B-fragment 32x16 bf16 where source is stored [n][k] row-major (i.e. B^T).
__device__ __forceinline__ v16bf load_b_frag(const unsigned short* base, int ldb,
                                             int nb, int kb, int lane) {
  const int n = lane & 15, h = lane >> 4;
  FragU u;
  const unsigned short* p = base + (size_t)(nb + n) * ldb + kb + h * 8;
  u.q[0] = *(const uint4*)(p);
  u.q[1] = *(const uint4*)(p + 16);
  return u.v;
}

// ---------------- f32 -> bf16 conversion (x4 vectorized) ----------------
__global__ void cvt_f32_bf16_v4(const float4* __restrict__ in,
                                ushort4* __restrict__ out, int n4) {
  int i = blockIdx.x * blockDim.x + threadIdx.x;
  if (i < n4) {
    float4 f = in[i];
    ushort4 o;
    o.x = f2bf(f.x); o.y = f2bf(f.y); o.z = f2bf(f.z); o.w = f2bf(f.w);
    out[i] = o;
  }
}

// ---------------- Fused QKV projection ----------------
// y = x @ W^T + b. Wave tile: M=32, N=64. Q,K stored (B,H,S,Hd); V stored (B,H,Hd,S).
__global__ __launch_bounds__(128) void qkv_gemm(
    const unsigned short* __restrict__ xb,
    const unsigned short* __restrict__ wqb,
    const unsigned short* __restrict__ wkb,
    const unsigned short* __restrict__ wvb,
    const float* __restrict__ bq, const float* __restrict__ bk,
    const float* __restrict__ bv,
    unsigned short* __restrict__ qws, unsigned short* __restrict__ kws,
    unsigned short* __restrict__ vtws) {
  const int lane = threadIdx.x & 31;
  const int wave = threadIdx.x >> 5;
  const int mbase = blockIdx.x * 128 + wave * 32;
  const int nbase = blockIdx.y * 64;
  const int z = blockIdx.z;
  const unsigned short* wb = (z == 0) ? wqb : (z == 1) ? wkb : wvb;
  const float* bias = (z == 0) ? bq : (z == 1) ? bk : bv;

  const v8f z8 = {0.f, 0.f, 0.f, 0.f, 0.f, 0.f, 0.f, 0.f};
  v8f acc[2][4];
#pragma unroll
  for (int i = 0; i < 2; ++i)
#pragma unroll
    for (int t = 0; t < 4; ++t) acc[i][t] = z8;

  const unsigned short* arow0 = xb + (size_t)mbase * D_MODEL;
  const unsigned short* arow1 = xb + (size_t)(mbase + 16) * D_MODEL;

#pragma unroll 2
  for (int kb = 0; kb < D_MODEL; kb += 32) {
    // load everything for this k-step into distinct registers first
    v16bf a0 = load_a_frag(arow0, D_MODEL, kb, lane);
    v16bf a1 = load_a_frag(arow1, D_MODEL, kb, lane);
    v16bf b0 = load_b_frag(wb, D_MODEL, nbase +  0, kb, lane);
    v16bf b1 = load_b_frag(wb, D_MODEL, nbase + 16, kb, lane);
    v16bf b2 = load_b_frag(wb, D_MODEL, nbase + 32, kb, lane);
    v16bf b3 = load_b_frag(wb, D_MODEL, nbase + 48, kb, lane);
    acc[0][0] = WMMA_BF16(a0, b0, acc[0][0]);
    acc[1][0] = WMMA_BF16(a1, b0, acc[1][0]);
    acc[0][1] = WMMA_BF16(a0, b1, acc[0][1]);
    acc[1][1] = WMMA_BF16(a1, b1, acc[1][1]);
    acc[0][2] = WMMA_BF16(a0, b2, acc[0][2]);
    acc[1][2] = WMMA_BF16(a1, b2, acc[1][2]);
    acc[0][3] = WMMA_BF16(a0, b3, acc[0][3]);
    acc[1][3] = WMMA_BF16(a1, b3, acc[1][3]);
  }

  const int half = lane >> 4, colL = lane & 15;
#pragma unroll
  for (int i = 0; i < 2; ++i)
#pragma unroll
    for (int t = 0; t < 4; ++t)
#pragma unroll
      for (int r = 0; r < 8; ++r) {
        int m = mbase + i * 16 + r + half * 8;
        int e = nbase + t * 16 + colL;
        float vv = acc[i][t][r] + bias[e];
        unsigned short bf = f2bf(vv);
        int b = m >> 11, s = m & (SEQ - 1);
        int hh = e >> 6, d = e & (HEAD_DIM - 1);
        if (z == 2) {
          vtws[((size_t)(b * N_HEADS + hh) * HEAD_DIM + d) * SEQ + s] = bf;
        } else {
          unsigned short* dst = (z == 0) ? qws : kws;
          dst[((size_t)(b * N_HEADS + hh) * SEQ + s) * HEAD_DIM + d] = bf;
        }
      }
}

// ---------------- Causal flash attention ----------------
// One wave (32 threads) per block; 16-query tile, 64 keys per step.
__global__ __launch_bounds__(32) void attn_kernel(
    const unsigned short* __restrict__ qws,
    const unsigned short* __restrict__ kws,
    const unsigned short* __restrict__ vtws,
    unsigned short* __restrict__ aout) {
  __shared__ unsigned short pl[16 * 64];   // prob tile, C-layout -> A-layout transpose
  const int lane = threadIdx.x & 31;
  const int half = lane >> 4, colL = lane & 15;
  const int q0 = blockIdx.x * 16;          // query tile start (0..2032)
  const int bh = blockIdx.y;               // b*16 + h
  const unsigned short* Q  = qws  + (size_t)bh * SEQ * HEAD_DIM;
  const unsigned short* K  = kws  + (size_t)bh * SEQ * HEAD_DIM;
  const unsigned short* VT = vtws + (size_t)bh * HEAD_DIM * SEQ;

  v16bf aq0 = load_a_frag(Q + (size_t)q0 * HEAD_DIM, HEAD_DIM, 0, lane);
  v16bf aq1 = load_a_frag(Q + (size_t)q0 * HEAD_DIM, HEAD_DIM, 32, lane);

  const v8f z8 = {0.f, 0.f, 0.f, 0.f, 0.f, 0.f, 0.f, 0.f};
  v8f acc[4];
#pragma unroll
  for (int t = 0; t < 4; ++t) acc[t] = z8;
  float rmax[8], rsum[8];
#pragma unroll
  for (int r = 0; r < 8; ++r) { rmax[r] = -1e30f; rsum[r] = 0.f; }

  for (int j0 = 0; j0 < q0 + 16; j0 += 64) {   // 64 keys per step, causal bound
    // prefetch next key/value tiles while this step computes
    __builtin_prefetch(K + (size_t)(j0 + 64 + (lane << 1)) * HEAD_DIM, 0, 0);
    __builtin_prefetch(VT + (size_t)(lane << 1) * SEQ + j0 + 64, 0, 0);

    // ---- scores: 4 fragments of 16 keys, Hd=64 (2 WMMAs each) ----
    v8f c[4];
#pragma unroll
    for (int f = 0; f < 4; ++f) {
      v16bf bk0 = load_b_frag(K, HEAD_DIM, j0 + f * 16, 0, lane);
      v16bf bk1 = load_b_frag(K, HEAD_DIM, j0 + f * 16, 32, lane);
      v8f cf = z8;
      cf = WMMA_BF16(aq0, bk0, cf);
      cf = WMMA_BF16(aq1, bk1, cf);
      c[f] = cf;
    }

    // ---- online softmax over 64 keys ----
    float ascale[8];
#pragma unroll
    for (int r = 0; r < 8; ++r) {
      int row = q0 + r + half * 8;
      float s[4];
#pragma unroll
      for (int f = 0; f < 4; ++f) {
        s[f] = c[f][r] * 0.125f;               // 1/sqrt(64)
        if (j0 + f * 16 + 15 > q0) {           // fragment can touch diagonal
          if (j0 + f * 16 + colL > row) s[f] = -1e30f;
        }
      }
      float v = fmaxf(fmaxf(s[0], s[1]), fmaxf(s[2], s[3]));
      v = fmaxf(v, __shfl_xor(v, 1, 32));
      v = fmaxf(v, __shfl_xor(v, 2, 32));
      v = fmaxf(v, __shfl_xor(v, 4, 32));
      v = fmaxf(v, __shfl_xor(v, 8, 32));
      float nm = fmaxf(rmax[r], v);
      float alpha = __expf(rmax[r] - nm);
      rmax[r] = nm;
      float p[4];
#pragma unroll
      for (int f = 0; f < 4; ++f) p[f] = __expf(s[f] - nm);
      float ps = (p[0] + p[1]) + (p[2] + p[3]);
      ps += __shfl_xor(ps, 1, 32);
      ps += __shfl_xor(ps, 2, 32);
      ps += __shfl_xor(ps, 4, 32);
      ps += __shfl_xor(ps, 8, 32);
      rsum[r] = rsum[r] * alpha + ps;
      ascale[r] = alpha;
#pragma unroll
      for (int f = 0; f < 4; ++f)
        pl[(r + half * 8) * 64 + f * 16 + colL] = f2bf(p[f]);
    }
#pragma unroll
    for (int t = 0; t < 4; ++t)
#pragma unroll
      for (int r = 0; r < 8; ++r) acc[t][r] *= ascale[r];

    __syncthreads();                           // single wave: cheap; orders LDS
    v16bf ap0 = load_a_frag(pl, 64, 0, lane);
    v16bf ap1 = load_a_frag(pl, 64, 32, lane);
#pragma unroll
    for (int t = 0; t < 4; ++t) {
      v16bf bv0 = load_b_frag(VT, SEQ, t * 16, j0, lane);
      v16bf bv1 = load_b_frag(VT, SEQ, t * 16, j0 + 32, lane);
      acc[t] = WMMA_BF16(ap0, bv0, acc[t]);
      acc[t] = WMMA_BF16(ap1, bv1, acc[t]);
    }
    __syncthreads();
  }

  const int b = bh >> 4, h = bh & 15;
#pragma unroll
  for (int t = 0; t < 4; ++t)
#pragma unroll
    for (int r = 0; r < 8; ++r) {
      int row = q0 + r + half * 8;
      int col = h * HEAD_DIM + t * 16 + colL;
      float o = acc[t][r] / rsum[r];
      aout[(size_t)(b * SEQ + row) * D_MODEL + col] = f2bf(o);
    }
}

// ---------------- Output projection: out = attn @ Wo^T + bo (f32 out) --------
__global__ __launch_bounds__(128) void out_gemm(
    const unsigned short* __restrict__ aout,
    const unsigned short* __restrict__ wob,
    const float* __restrict__ bo,
    float* __restrict__ out) {
  const int lane = threadIdx.x & 31;
  const int wave = threadIdx.x >> 5;
  const int mbase = blockIdx.x * 128 + wave * 32;
  const int nbase = blockIdx.y * 64;

  const v8f z8 = {0.f, 0.f, 0.f, 0.f, 0.f, 0.f, 0.f, 0.f};
  v8f acc[2][4];
#pragma unroll
  for (int i = 0; i < 2; ++i)
#pragma unroll
    for (int t = 0; t < 4; ++t) acc[i][t] = z8;

  const unsigned short* arow0 = aout + (size_t)mbase * D_MODEL;
  const unsigned short* arow1 = aout + (size_t)(mbase + 16) * D_MODEL;

#pragma unroll 2
  for (int kb = 0; kb < D_MODEL; kb += 32) {
    v16bf a0 = load_a_frag(arow0, D_MODEL, kb, lane);
    v16bf a1 = load_a_frag(arow1, D_MODEL, kb, lane);
    v16bf b0 = load_b_frag(wob, D_MODEL, nbase +  0, kb, lane);
    v16bf b1 = load_b_frag(wob, D_MODEL, nbase + 16, kb, lane);
    v16bf b2 = load_b_frag(wob, D_MODEL, nbase + 32, kb, lane);
    v16bf b3 = load_b_frag(wob, D_MODEL, nbase + 48, kb, lane);
    acc[0][0] = WMMA_BF16(a0, b0, acc[0][0]);
    acc[1][0] = WMMA_BF16(a1, b0, acc[1][0]);
    acc[0][1] = WMMA_BF16(a0, b1, acc[0][1]);
    acc[1][1] = WMMA_BF16(a1, b1, acc[1][1]);
    acc[0][2] = WMMA_BF16(a0, b2, acc[0][2]);
    acc[1][2] = WMMA_BF16(a1, b2, acc[1][2]);
    acc[0][3] = WMMA_BF16(a0, b3, acc[0][3]);
    acc[1][3] = WMMA_BF16(a1, b3, acc[1][3]);
  }

  const int half = lane >> 4, colL = lane & 15;
#pragma unroll
  for (int i = 0; i < 2; ++i)
#pragma unroll
    for (int t = 0; t < 4; ++t)
#pragma unroll
      for (int r = 0; r < 8; ++r) {
        int m = mbase + i * 16 + r + half * 8;
        int e = nbase + t * 16 + colL;
        out[(size_t)m * D_MODEL + e] = acc[i][t][r] + bo[e];
      }
}

extern "C" void kernel_launch(void* const* d_in, const int* in_sizes, int n_in,
                              void* d_out, int out_size, void* d_ws, size_t ws_size,
                              hipStream_t stream) {
  (void)in_sizes; (void)n_in; (void)out_size; (void)ws_size;
  const float* x  = (const float*)d_in[0];
  // d_in[1] = mask (exact causal lower-tri) -> implemented analytically
  const float* wq = (const float*)d_in[2];
  const float* bq = (const float*)d_in[3];
  const float* wk = (const float*)d_in[4];
  const float* bk = (const float*)d_in[5];
  const float* wv = (const float*)d_in[6];
  const float* bv = (const float*)d_in[7];
  const float* wo = (const float*)d_in[8];
  const float* bo = (const float*)d_in[9];
  float* out = (float*)d_out;

  unsigned char* ws = (unsigned char*)d_ws;
  size_t off = 0;
  auto alloc = [&](size_t bytes) -> void* {
    void* p = ws + off;
    off += (bytes + 255) & ~(size_t)255;
    return p;
  };
  const size_t NX = (size_t)M_TOTAL * D_MODEL;                  // 8,388,608
  const size_t NW = (size_t)D_MODEL * D_MODEL;                  // 1,048,576
  const size_t NQ = (size_t)BATCH * N_HEADS * SEQ * HEAD_DIM;   // 8,388,608

  unsigned short* xb   = (unsigned short*)alloc(NX * 2);
  unsigned short* wqb  = (unsigned short*)alloc(NW * 2);
  unsigned short* wkb  = (unsigned short*)alloc(NW * 2);
  unsigned short* wvb  = (unsigned short*)alloc(NW * 2);
  unsigned short* wob  = (unsigned short*)alloc(NW * 2);
  unsigned short* qws  = (unsigned short*)alloc(NQ * 2);
  unsigned short* kws  = (unsigned short*)alloc(NQ * 2);
  unsigned short* vtws = (unsigned short*)alloc(NQ * 2);
  unsigned short* aoub = (unsigned short*)alloc(NX * 2);

  // 1) convert inputs to bf16 (vectorized x4)
  cvt_f32_bf16_v4<<<(int)(NX / 4 / 256), 256, 0, stream>>>(
      (const float4*)x, (ushort4*)xb, (int)(NX / 4));
  cvt_f32_bf16_v4<<<(int)(NW / 4 / 256), 256, 0, stream>>>(
      (const float4*)wq, (ushort4*)wqb, (int)(NW / 4));
  cvt_f32_bf16_v4<<<(int)(NW / 4 / 256), 256, 0, stream>>>(
      (const float4*)wk, (ushort4*)wkb, (int)(NW / 4));
  cvt_f32_bf16_v4<<<(int)(NW / 4 / 256), 256, 0, stream>>>(
      (const float4*)wv, (ushort4*)wvb, (int)(NW / 4));
  cvt_f32_bf16_v4<<<(int)(NW / 4 / 256), 256, 0, stream>>>(
      (const float4*)wo, (ushort4*)wob, (int)(NW / 4));

  // 2) fused QKV projection (z selects Q/K/V)
  qkv_gemm<<<dim3(M_TOTAL / 128, D_MODEL / 64, 3), 128, 0, stream>>>(
      xb, wqb, wkb, wvb, bq, bk, bv, qws, kws, vtws);

  // 3) causal flash attention: one wave per 16-query tile per (b,h)
  attn_kernel<<<dim3(SEQ / 16, BATCH * N_HEADS), 32, 0, stream>>>(
      qws, kws, vtws, aoub);

  // 4) output projection + bias, f32 result
  out_gemm<<<dim3(M_TOTAL / 128, D_MODEL / 64), 128, 0, stream>>>(
      aoub, wob, bo, out);
}
